// Net_85864986182131
// MI455X (gfx1250) — matrix operand order repaired and checked by reading, not compiled, not run on hardware
//
#include <hip/hip_runtime.h>
#include <hip/hip_bf16.h>

typedef __attribute__((ext_vector_type(16))) _Float16 v16h;
typedef __attribute__((ext_vector_type(8)))  _Float16 v8h;
typedef __attribute__((ext_vector_type(8)))  float    v8f;

union V16U { v16h v; v8h h[2]; };

// ---------------------------------------------------------------------------
// WMMA helpers (CDNA5 wave32 layouts, CDNA5 ISA §7.12.2)
// A 16x32 f16: lane l -> row r=l&15, hs=l>>4; element e -> k = 16*hs+e
//              (matches B).  => two contiguous 8-half chunks.
// ---------------------------------------------------------------------------
__device__ inline v16h wmma_load_a_row(const _Float16* __restrict__ row, int hs) {
    V16U u;
    u.h[0] = *(const v8h*)(row + 16 * hs);
    u.h[1] = *(const v8h*)(row + 8 + 16 * hs);
    return u.v;
}
// B 32x16 f16 from Bt[N][K] row-major: lane l -> col n=l&15, hs=l>>4;
// element e -> k = 16*hs + e  => one contiguous 16-half chunk of row n.
__device__ inline v16h wmma_load_b_row(const _Float16* __restrict__ nrow, int hs) {
    return *(const v16h*)(nrow + 16 * hs);
}

__device__ inline v8f wmma_acc(v16h a, v16h b, v8f c) {
    return __builtin_amdgcn_wmma_f32_16x16x32_f16(false, a, false, b, (short)0, c,
                                                  false, false);
}

// ---------------------------------------------------------------------------
// Workspace zero-fill (for the padded spike buffer halo), 16B per thread.
// ---------------------------------------------------------------------------
__global__ __launch_bounds__(256) void k_zero16(uint4* __restrict__ p, int n16) {
    int i = blockIdx.x * 256 + threadIdx.x;
    if (i < n16) p[i] = make_uint4(0u, 0u, 0u, 0u);
}

// ---------------------------------------------------------------------------
// Weight conversion kernels
// ---------------------------------------------------------------------------
__global__ __launch_bounds__(256) void k_cvt_f16(const float* __restrict__ src,
                                                 _Float16* __restrict__ dst, int n) {
    int i = blockIdx.x * 256 + threadIdx.x;
    if (i < n) dst[i] = (_Float16)src[i];
}

// conv2_w [O=128][I=128][3][3] f32 -> wt [tap=kh*3+kw][O][I] f16
__global__ __launch_bounds__(256) void k_retap_conv2(const float* __restrict__ w,
                                                     _Float16* __restrict__ wt) {
    int i = blockIdx.x * 256 + threadIdx.x;            // 9*128*128 = 147456
    if (i >= 147456) return;
    int ic   = i & 127;
    int rest = i >> 7;
    int oc   = rest & 127;
    int tap  = rest >> 7;
    wt[i] = (_Float16)w[(size_t)(oc * 128 + ic) * 9 + tap];
}

// ---------------------------------------------------------------------------
// Stage A: conv1(1->128, 3x3 SAME, 28x28) + BN1 + IF scan (T=8) + maxpool 2x2
// Output p1: f16 spikes in a ZERO-PADDED NHWC buffer [TB][16][16][128]
// (interior rows 1..14, cols 1..14; halo pre-zeroed by k_zero16) so conv2's
// implicit-GEMM A loads never need boundary predicates.
// ---------------------------------------------------------------------------
__global__ __launch_bounds__(256) void k_conv1_if_pool(
    const float* __restrict__ x,      // [64,1,28,28]
    const float* __restrict__ w1,     // [128,1,3,3]
    const float* __restrict__ g, const float* __restrict__ bb,
    const float* __restrict__ mm, const float* __restrict__ vv,
    _Float16* __restrict__ p1) {
    int tid = blockIdx.x * 256 + threadIdx.x;          // 64*196*128 = 1605632
    int c    = tid & 127;
    int rest = tid >> 7;                               // b*196 + p
    int p    = rest % 196;
    int b    = rest / 196;
    int oy = p / 14, ox = p % 14;

    float w[9];
#pragma unroll
    for (int j = 0; j < 9; ++j) w[j] = w1[c * 9 + j];
    float sc = g[c] * rsqrtf(vv[c] + 1e-5f);
    float bi = bb[c] - mm[c] * sc;

    const float* xb = x + (size_t)b * 784;
    float y0[4];
#pragma unroll
    for (int i = 0; i < 4; ++i) {
        int py = oy * 2 + (i >> 1);
        int px = ox * 2 + (i & 1);
        float s = 0.f;
#pragma unroll
        for (int u = 0; u < 3; ++u) {
            int yy = py + u - 1;
            if ((unsigned)yy >= 28u) continue;
#pragma unroll
            for (int v = 0; v < 3; ++v) {
                int xx = px + v - 1;
                if ((unsigned)xx >= 28u) continue;
                s += xb[yy * 28 + xx] * w[u * 3 + v];
            }
        }
        y0[i] = s * sc + bi;
    }

    float vm[4] = {0.f, 0.f, 0.f, 0.f};
#pragma unroll
    for (int t = 0; t < 8; ++t) {
        float pooled = 0.f;
#pragma unroll
        for (int i = 0; i < 4; ++i) {
            vm[i] += y0[i];
            float s = (vm[i] >= 1.0f) ? 1.0f : 0.0f;   // spike(v - V_TH)
            vm[i] = (s > 0.f) ? 0.f : vm[i];           // hard reset
            pooled = fmaxf(pooled, s);
        }
        // padded: [(t*64+b)][oy+1][ox+1][c] with H=W=16
        p1[((size_t)(t * 64 + b) * 256 + (oy + 1) * 16 + (ox + 1)) * 128 + c] =
            (_Float16)pooled;
    }
}

// ---------------------------------------------------------------------------
// Stage B: conv2 as implicit GEMM (9 tap-GEMMs) + BN2 fold.
// M = 512*196 rows (tb, oh, ow), N = 128, K = 9*128. Wave tile = 16x16.
// Branch-free (padded input), manually double-buffered K pipeline.
// ---------------------------------------------------------------------------
__device__ inline const _Float16* conv2_aptr(const _Float16* __restrict__ p1,
                                             int tb, int oh, int ow, int s) {
    const int tap = s >> 2;
    const int kk  = (s & 3) * 32;
    const int kh  = tap / 3;
    const int kw  = tap - kh * 3;
    // padded coords: ih = oh + kh, iw = ow + kw  (both in [0,16))
    return p1 + ((size_t)tb * 256 + (oh + kh) * 16 + (ow + kw)) * 128 + kk;
}
__device__ inline const _Float16* conv2_bptr(const _Float16* __restrict__ wt,
                                             int n, int s) {
    const int tap = s >> 2;
    const int kk  = (s & 3) * 32;
    return wt + ((size_t)tap * 128 + n) * 128 + kk;
}

__global__ __launch_bounds__(256) void k_conv2_wmma(
    const _Float16* __restrict__ p1,   // [512][16][16][128] padded spikes
    const _Float16* __restrict__ wt,   // [9][128][128] f16 tap-major weights
    const float* __restrict__ g2, const float* __restrict__ b2,
    const float* __restrict__ m2, const float* __restrict__ v2,
    float* __restrict__ y2) {          // [512*196, 128] f32
    const int lane = threadIdx.x & 31;
    const int wid  = threadIdx.x >> 5;
    const int tile = blockIdx.x * 8 + wid;             // 6272 * 8 = 50176 tiles
    const int mt = tile >> 3;                          // N tiles = 128/16 = 8
    const int nt = tile & 7;
    const int r  = lane & 15;
    const int hs = lane >> 4;

    const int m   = mt * 16 + r;                       // this lane's A row
    const int tb  = m / 196;
    const int pix = m % 196;
    const int oh  = pix / 14;
    const int ow  = pix % 14;
    const int n   = nt * 16 + r;                       // this lane's B column

    v8f acc = {};
    v16h a = wmma_load_a_row(conv2_aptr(p1, tb, oh, ow, 0), hs);
    v16h b = wmma_load_b_row(conv2_bptr(wt, n, 0), hs);
#pragma unroll
    for (int s = 0; s < 36; ++s) {
        v16h an = a, bn = b;
        if (s < 35) {                                  // constant under unroll
            an = wmma_load_a_row(conv2_aptr(p1, tb, oh, ow, s + 1), hs);
            bn = wmma_load_b_row(conv2_bptr(wt, n, s + 1), hs);
        }
        acc = wmma_acc(a, b, acc);
        a = an;
        b = bn;
    }

    // BN2 epilogue (per output channel n, fixed per lane)
    float sc = g2[n] * rsqrtf(v2[n] + 1e-5f);
    float bi = b2[n] - m2[n] * sc;
    const int mbase = mt * 16 + 8 * hs;                // D: row = v + 8*hs
#pragma unroll
    for (int v = 0; v < 8; ++v)
        y2[(size_t)(mbase + v) * 128 + n] = acc[v] * sc + bi;
}

// ---------------------------------------------------------------------------
// Stage C: IF scan over T on conv2 output + maxpool 2x2.
// Writes p2 in NCHW-flatten order (c*49 + oy*7+ox) per (t,b) row so fc1_w
// columns line up: p2 row = t*64+b, K = 6272.
// ---------------------------------------------------------------------------
__global__ __launch_bounds__(256) void k_if_pool2(const float* __restrict__ y2,
                                                  _Float16* __restrict__ p2) {
    int tid = blockIdx.x * 256 + threadIdx.x;          // 64*49*128 = 401408
    int c    = tid & 127;
    int rest = tid >> 7;                               // b*49 + p
    int p    = rest % 49;
    int b    = rest / 49;
    int oy = p / 7, ox = p % 7;

    float vm[4] = {0.f, 0.f, 0.f, 0.f};
#pragma unroll
    for (int t = 0; t < 8; ++t) {
        float pooled = 0.f;
#pragma unroll
        for (int i = 0; i < 4; ++i) {
            int iy = oy * 2 + (i >> 1);
            int ix = ox * 2 + (i & 1);
            float h = y2[((size_t)(t * 64 + b) * 196 + iy * 14 + ix) * 128 + c];
            vm[i] += h;
            float s = (vm[i] >= 1.0f) ? 1.0f : 0.0f;
            vm[i] = (s > 0.f) ? 0.f : vm[i];
            pooled = fmaxf(pooled, s);
        }
        p2[(size_t)(t * 64 + b) * 6272 + c * 49 + p] = (_Float16)pooled;
    }
}

// ---------------------------------------------------------------------------
// Generic wave-tile WMMA GEMM: C[M,N] = A[M,K] * Bt[N,K]^T  (f16 in, f32 out)
// M, N multiples of 16; K multiple of 32. Double-buffered K loop + prefetch.
// ---------------------------------------------------------------------------
__global__ __launch_bounds__(256) void k_gemm_f16(const _Float16* __restrict__ A,
                                                  const _Float16* __restrict__ Bt,
                                                  float* __restrict__ C,
                                                  int M, int N, int K) {
    const int lane = threadIdx.x & 31;
    const int wid  = threadIdx.x >> 5;
    const int tile = blockIdx.x * 8 + wid;
    const int ntiles = N >> 4;
    const int total  = (M >> 4) * ntiles;
    if (tile >= total) return;                         // wave-uniform exit
    const int mt = tile / ntiles;
    const int nt = tile % ntiles;
    const int r  = lane & 15;
    const int hs = lane >> 4;

    const _Float16* arow = A + (size_t)(mt * 16 + r) * K;
    const _Float16* brow = Bt + (size_t)(nt * 16 + r) * K;

    v8f acc = {};
    v16h a = wmma_load_a_row(arow, hs);
    v16h b = wmma_load_b_row(brow, hs);
    for (int kk = 32; kk < K; kk += 32) {
        v16h an = wmma_load_a_row(arow + kk, hs);
        v16h bn = wmma_load_b_row(brow + kk, hs);
        __builtin_prefetch(arow + kk + 256, 0, 1);     // global_prefetch_b8
        __builtin_prefetch(brow + kk + 256, 0, 1);
        acc = wmma_acc(a, b, acc);
        a = an;
        b = bn;
    }
    acc = wmma_acc(a, b, acc);

    const int mbase = mt * 16 + 8 * hs;
    const int n = nt * 16 + r;
#pragma unroll
    for (int v = 0; v < 8; ++v)
        C[(size_t)(mbase + v) * N + n] = acc[v];
}

// ---------------------------------------------------------------------------
// LIF scan over T: v = 0.5*(v + x); spike at v>=1; hard reset. s out as f16.
// Rows are (t*64+b), columns j in [0,N).
// ---------------------------------------------------------------------------
__global__ __launch_bounds__(256) void k_lif(const float* __restrict__ h,
                                             _Float16* __restrict__ s, int N) {
    int tid = blockIdx.x * 256 + threadIdx.x;          // 64*N threads
    int j = tid % N;
    int b = tid / N;
    float v = 0.f;
#pragma unroll
    for (int t = 0; t < 8; ++t) {
        float x = h[(size_t)(t * 64 + b) * N + j];
        v = 0.5f * (v + x);                            // v += (x - v)/tau, tau=2
        float sp = (v >= 1.0f) ? 1.0f : 0.0f;
        v = (sp > 0.f) ? 0.f : v;
        s[(size_t)(t * 64 + b) * N + j] = (_Float16)sp;
    }
}

// ---------------------------------------------------------------------------
// Tail: fc3 (10x128) + LIF + temporal mean -> out[64,10] f32
// ---------------------------------------------------------------------------
__global__ __launch_bounds__(256) void k_fc3_head(const _Float16* __restrict__ s4,
                                                  const float* __restrict__ w3,
                                                  float* __restrict__ out) {
    int tid = blockIdx.x * 256 + threadIdx.x;
    if (tid >= 640) return;
    int o = tid % 10;
    int b = tid / 10;
    const float* wrow = w3 + (size_t)o * 128;
    float v = 0.f, acc = 0.f;
#pragma unroll
    for (int t = 0; t < 8; ++t) {
        const _Float16* srow = s4 + (size_t)(t * 64 + b) * 128;
        float h = 0.f;
#pragma unroll
        for (int k = 0; k < 128; ++k) h += (float)srow[k] * wrow[k];
        v = 0.5f * (v + h);
        float sp = (v >= 1.0f) ? 1.0f : 0.0f;
        v = (sp > 0.f) ? 0.f : v;
        acc += sp;
    }
    out[b * 10 + o] = acc * 0.125f;
}

// ---------------------------------------------------------------------------
extern "C" void kernel_launch(void* const* d_in, const int* in_sizes, int n_in,
                              void* d_out, int out_size, void* d_ws, size_t ws_size,
                              hipStream_t stream) {
    const float* x       = (const float*)d_in[0];   // [64,1,28,28]
    const float* conv1_w = (const float*)d_in[1];   // [128,1,3,3]
    const float* bn1_g   = (const float*)d_in[2];
    const float* bn1_b   = (const float*)d_in[3];
    const float* bn1_m   = (const float*)d_in[4];
    const float* bn1_v   = (const float*)d_in[5];
    const float* conv2_w = (const float*)d_in[6];   // [128,128,3,3]
    const float* bn2_g   = (const float*)d_in[7];
    const float* bn2_b   = (const float*)d_in[8];
    const float* bn2_m   = (const float*)d_in[9];
    const float* bn2_v   = (const float*)d_in[10];
    const float* fc1_w   = (const float*)d_in[11];  // [1152, 6272]
    const float* fc2_w   = (const float*)d_in[12];  // [128, 1152]
    const float* fc3_w   = (const float*)d_in[13];  // [10, 128]
    float* out = (float*)d_out;

    // ---- workspace carve-up (256B aligned) ----
    char* ws = (char*)d_ws;
    size_t off = 0;
    auto take = [&](size_t bytes) {
        char* p = ws + off;
        off = (off + bytes + 255) & ~(size_t)255;
        return p;
    };
    const size_t P1_HALVES = (size_t)512 * 16 * 16 * 128;           // padded spikes
    _Float16* p1   = (_Float16*)take(P1_HALVES * 2);
    _Float16* wt   = (_Float16*)take((size_t)9 * 128 * 128 * 2);    // conv2 tap weights
    _Float16* fc1h = (_Float16*)take((size_t)1152 * 6272 * 2);
    _Float16* fc2h = (_Float16*)take((size_t)128 * 1152 * 2);
    float*    y2   = (float*)take((size_t)512 * 196 * 128 * 4);     // conv2+BN2 out
    _Float16* p2   = (_Float16*)take((size_t)512 * 6272 * 2);       // spikes L2 pooled
    float*    h1   = (float*)take((size_t)512 * 1152 * 4);
    _Float16* s3   = (_Float16*)take((size_t)512 * 1152 * 2);
    float*    h2   = (float*)take((size_t)512 * 128 * 4);
    _Float16* s4   = (_Float16*)take((size_t)512 * 128 * 2);
    (void)ws_size; (void)in_sizes; (void)n_in; (void)out_size;

    // ---- clear padded spike buffer (halo must be exact zeros every call) ----
    const int n16 = (int)(P1_HALVES * 2 / 16);                      // 2,097,152
    k_zero16<<<(n16 + 255) / 256, 256, 0, stream>>>((uint4*)p1, n16);

    // ---- weight prep ----
    k_retap_conv2<<<576, 256, 0, stream>>>(conv2_w, wt);
    k_cvt_f16<<<(1152 * 6272 + 255) / 256, 256, 0, stream>>>(fc1_w, fc1h, 1152 * 6272);
    k_cvt_f16<<<(128 * 1152 + 255) / 256, 256, 0, stream>>>(fc2_w, fc2h, 128 * 1152);

    // ---- pipeline ----
    k_conv1_if_pool<<<6272, 256, 0, stream>>>(x, conv1_w, bn1_g, bn1_b, bn1_m,
                                              bn1_v, p1);
    k_conv2_wmma<<<6272, 256, 0, stream>>>(p1, wt, bn2_g, bn2_b, bn2_m, bn2_v, y2);
    k_if_pool2<<<1568, 256, 0, stream>>>(y2, p2);

    // fc1: M=512, N=1152, K=6272 -> 2304 wave tiles -> 288 blocks
    k_gemm_f16<<<288, 256, 0, stream>>>(p2, fc1h, h1, 512, 1152, 6272);
    k_lif<<<288, 256, 0, stream>>>(h1, s3, 1152);

    // fc2: M=512, N=128, K=1152 -> 256 wave tiles -> 32 blocks
    k_gemm_f16<<<32, 256, 0, stream>>>(s3, fc2h, h2, 512, 128, 1152);
    k_lif<<<32, 256, 0, stream>>>(h2, s4, 128);

    k_fc3_head<<<3, 256, 0, stream>>>(s4, fc3_w, out);
}
